// HyperConnections_37056977830027
// MI455X (gfx1250) — matrix operand order repaired and checked by reading, not compiled, not run on hardware
//
#include <hip/hip_runtime.h>
#include <math.h>

// ---------------- problem constants (from reference) ----------------
#define STREAMS 4
#define DIM     1024
#define SEQ     4096
#define BATCH   4
#define TB      8         // tokens per block; WMMA M = 2 streams x 8 tokens = 16
#define BLOCK   128       // 4 wave32s

// ---------------- CDNA5 vector types ----------------
typedef _Float16 v16h __attribute__((ext_vector_type(16)));
typedef _Float16 v8h  __attribute__((ext_vector_type(8)));
typedef float    v8f  __attribute__((ext_vector_type(8)));
typedef float    v4f  __attribute__((ext_vector_type(4)));
typedef unsigned int u32x4 __attribute__((ext_vector_type(4)));
typedef int      i32x4 __attribute__((ext_vector_type(4)));
typedef int      i32x8 __attribute__((ext_vector_type(8)));

// ---------------- LDS layout (bytes) ----------------
//  R tile  : 4 streams x 8 tokens x 1024 f32  = 131072
//  Wgh     : 9 cols x 1024 f16 (col 8 = zeros)=  18432
//  dots    : 4 x 8 x 6 f32                    =    768
//  sumsq   : 4 x 8 f32                        =    128
//  coeff   : 8 x 4 x 4 f32                    =    512
//  total 150912 B  ->  TWO workgroups co-resident per 320KB WGP, so the
//  scheduler overlaps one block's TDM load with the other's compute/store.
#define LDS_R_OFF    0
#define LDS_W_OFF    131072
#define LDS_DOTS_OFF 149504
#define LDS_SQ_OFF   150272
#define LDS_C_OFF    150400
#define LDS_TOTAL    150912

__device__ __forceinline__ v16h cvt4x4_to_v16h(v4f f0, v4f f1, v4f f2, v4f f3) {
  v16h a;
  a[0]=(_Float16)f0.x; a[1]=(_Float16)f0.y; a[2] =(_Float16)f0.z; a[3] =(_Float16)f0.w;
  a[4]=(_Float16)f1.x; a[5]=(_Float16)f1.y; a[6] =(_Float16)f1.z; a[7] =(_Float16)f1.w;
  a[8]=(_Float16)f2.x; a[9]=(_Float16)f2.y; a[10]=(_Float16)f2.z; a[11]=(_Float16)f2.w;
  a[12]=(_Float16)f3.x;a[13]=(_Float16)f3.y;a[14]=(_Float16)f3.z; a[15]=(_Float16)f3.w;
  return a;
}

extern "C" __global__ __launch_bounds__(BLOCK)
void hyperconn_kernel(const float* __restrict__ resid,
                      const float* __restrict__ gamma,
                      const float* __restrict__ stA,      // [4][5]
                      const float* __restrict__ aFn,      // [1024][5]
                      const float* __restrict__ aScale,   // scalar
                      const float* __restrict__ stB,      // [4]
                      const float* __restrict__ bFn,      // [1024]
                      const float* __restrict__ bScale,   // scalar
                      float* __restrict__ out)
{
  extern __shared__ char smem[];
  float*    smR    = (float*)(smem + LDS_R_OFF);      // [s][m][d]  (s*TB+m major)
  _Float16* smW    = (_Float16*)(smem + LDS_W_OFF);   // [col 0..8][d]
  float*    smDots = (float*)(smem + LDS_DOTS_OFF);   // [s*TB+m][t 0..5]
  float*    smSq   = (float*)(smem + LDS_SQ_OFF);     // [s*TB+m]
  float*    smC    = (float*)(smem + LDS_C_OFF);      // [m][sout][sin]

  const int tid  = threadIdx.x;
  const int wid  = tid >> 5;
  const int lane = tid & 31;

  const int t0 = blockIdx.x * TB;          // first token of this block
  const int bb = t0 / SEQ;                 // batch index (tokens never straddle batches)
  const int n0 = t0 % SEQ;

  // ---- Phase A: wave 0 issues ONE 3-D TDM tile load: x=1024 dims, y=8 tokens,
  //      z=4 streams (dim1_stride = SEQ*DIM = stream stride). LDS fill order
  //      is x-major then y then z => exactly the [s][m][d] layout above.
  if (wid == 0) {
    unsigned ldsR0 = (unsigned)(unsigned long long)(void*)smR; // low 32b = LDS offset
    unsigned long long ga =
      (unsigned long long)(const void*)(resid + ((size_t)bb*STREAMS*SEQ + n0)*DIM);
    u32x4 g0; i32x8 g1; i32x4 g2; i32x4 g3 = {0,0,0,0};
    g0[0] = 1u;                                               // count=1, user descriptor
    g0[1] = ldsR0;                                            // lds_addr
    g0[2] = (unsigned)(ga & 0xFFFFFFFFull);                   // global_addr[31:0]
    g0[3] = (unsigned)((ga >> 32) & 0x1FFFFFFull) | (2u<<30); // global_addr[56:32] | type=2
    g1[0] = (int)(2u << 16);          // data_size = 4 bytes
    g1[1] = (int)(1024u << 16);       // tensor_dim0 = 1024   (lo16 @ bit48)
    g1[2] = (int)(4096u << 16);       // dim0 hi=0 | tensor_dim1 = 4096 lo16
    g1[3] = (int)(1024u << 16);       // dim1 hi=0 | tile_dim0 = 1024
    g1[4] = (int)(TB | (4u << 16));   // tile_dim1 = 8 tokens, tile_dim2 = 4 streams
    g1[5] = (int)1024u;               // tensor_dim0_stride = 1024 elements
    g1[6] = 0;                        // dim0_stride hi | dim1_stride lo16 (0x400000 lo16 = 0)
    g1[7] = (int)(0x400000u >> 16);   // dim1_stride = SEQ*DIM = 4194304 elements
    g2[0] = (int)4u;                  // tensor_dim2 = 4 streams
    g2[1] = 0; g2[2] = 0; g2[3] = 0;  // no dim3, no iterate
#if defined(__clang_major__) && (__clang_major__ >= 23)
    i32x8 g4 = {0,0,0,0,0,0,0,0};
    __builtin_amdgcn_tensor_load_to_lds(g0, g1, g2, g3, g4, 0);
#else
    __builtin_amdgcn_tensor_load_to_lds(g0, g1, g2, g3, 0);
#endif
  }

  // ---- Phase B (overlaps TDM): f16 weight tile Wgh[col][d] = (gamma+1)*W;
  //      cols 0..4 alpha_fn, col 5 beta_fn, cols 6..8 zeros (B-matrix padding).
  for (int d = tid; d < DIM; d += BLOCK) {
    float g = gamma[d] + 1.0f;
    smW[0*DIM + d] = (_Float16)(g * aFn[d*5 + 0]);
    smW[1*DIM + d] = (_Float16)(g * aFn[d*5 + 1]);
    smW[2*DIM + d] = (_Float16)(g * aFn[d*5 + 2]);
    smW[3*DIM + d] = (_Float16)(g * aFn[d*5 + 3]);
    smW[4*DIM + d] = (_Float16)(g * aFn[d*5 + 4]);
    smW[5*DIM + d] = (_Float16)(g * bFn[d]);
    smW[6*DIM + d] = (_Float16)0.0f;
    smW[7*DIM + d] = (_Float16)0.0f;
    smW[8*DIM + d] = (_Float16)0.0f;
  }

  if (wid == 0) __builtin_amdgcn_s_wait_tensorcnt(0);
  __syncthreads();

  // ---- Phase C: waves 0-1 -> WMMA reductions (M=16 = 2 streams x 8 tokens);
  //               waves 2-3 -> sum-of-squares with VALU + shfl.
  if (wid < 2) {
    const int mrow = lane & 15;     // A-matrix row = (stream pair local row)
    const int hi   = lane >> 4;     // K-half selector
    // A row mrow of wave w covers stream 2w + (mrow>>3), token mrow&7:
    // flat LDS row index = w*16 + mrow.
    const float* rRow = smR + (wid*16 + mrow)*DIM;
    // B column n = lane&15; lanes n>=8 read the dedicated zero column 8.
    const int n = lane & 15;
    const _Float16* wCol = smW + (n < 8 ? n : 8)*DIM;
    v8f acc = {0.f,0.f,0.f,0.f,0.f,0.f,0.f,0.f};
    #pragma unroll 4
    for (int ks = 0; ks < DIM/32; ++ks) {
      // A (16x32 f16): lane hi=0 holds K 0-7 & 16-23 of the 32-slice, hi=1 holds +8.
      const int ka = ks*32 + hi*8;
      v4f f0 = *(const v4f*)(rRow + ka);
      v4f f1 = *(const v4f*)(rRow + ka + 4);
      v4f f2 = *(const v4f*)(rRow + ka + 16);
      v4f f3 = *(const v4f*)(rRow + ka + 20);
      v16h a = cvt4x4_to_v16h(f0, f1, f2, f3);
      // B (32x16 f16): lane hi=0 holds K 0-7 & 16-23 of the 32-slice, hi=1 holds +8 (matches A).
      const int kb = ks*32 + hi*8;
      v8h b0 = *(const v8h*)(wCol + kb);
      v8h b1 = *(const v8h*)(wCol + kb + 16);
      v16h b = __builtin_shufflevector(b0, b1, 0,1,2,3,4,5,6,7,8,9,10,11,12,13,14,15);
      acc = __builtin_amdgcn_wmma_f32_16x16x32_f16(false, a, false, b,
                                                   (short)0, acc, false, false);
    }
    // D layout: acc[j] = (M = j + 8*hi, N = lane&15).
    // M maps to stream 2*wid + hi, token j.
    if (n < 6) {
      const int srow = (2*wid + hi)*TB;
      #pragma unroll
      for (int j = 0; j < 8; ++j)
        smDots[(srow + j)*6 + n] = acc[j];
    }
  } else {
    // 32 (stream,token) pairs over 2 waves; 2 lanes per pair, 512 elems per lane.
    const int p = (wid - 2)*16 + (lane & 15);   // flat row index s*TB+m
    const int h = lane >> 4;
    const float* rRow = smR + p*DIM + h*(DIM/2);
    float ss = 0.0f;
    for (int i = 0; i < DIM/2; i += 4) {
      v4f v = *(const v4f*)(rRow + i);
      ss += v.x*v.x + v.y*v.y + v.z*v.z + v.w*v.w;
    }
    ss += __shfl_xor(ss, 16, 32);
    if (h == 0) smSq[p] = ss;
  }
  __syncthreads();

  // ---- Phase D: 32 threads build per-token 4x4 mixing matrices.
  // coeff[sout][sin] = alpha[sin][sout+1] + beta[sout]*alpha[sin][0]
  if (tid < TB*STREAMS) {
    const int m  = tid >> 2;
    const int so = tid & 3;
    const float asc = aScale[0];
    const float bsc = bScale[0];
    float sc[4], a0[4], at[4];
    #pragma unroll
    for (int si = 0; si < 4; ++si) {
      const int row = si*TB + m;
      float nrm = sqrtf(smSq[row]);
      sc[si] = 32.0f / fmaxf(nrm, 1e-12f);      // sqrt(DIM)/max(||r||,eps)
      a0[si] = tanhf(smDots[row*6 + 0]      * sc[si]) * asc + stA[si*5 + 0];
      at[si] = tanhf(smDots[row*6 + (so+1)] * sc[si]) * asc + stA[si*5 + (so+1)];
    }
    float beta = tanhf(smDots[(so*TB + m)*6 + 5] * sc[so]) * bsc + stB[so];
    #pragma unroll
    for (int si = 0; si < 4; ++si)
      smC[(m*4 + so)*4 + si] = at[si] + beta * a0[si];
  }
  __syncthreads();

  // ---- Phase E: mix & stream out; nontemporal b128 stores (no reuse). ----
  for (int it = 0; it < STREAMS*TB*2; ++it) {
    const int row  = it >> 1;            // output row: s = row>>3, m = row&7
    const int half = it & 1;
    const int s = row >> 3;
    const int m = row & 7;
    const int d = (tid + half*BLOCK)*4;
    const float* cp = smC + (m*4 + s)*4;
    const float c0 = cp[0], c1 = cp[1], c2 = cp[2], c3 = cp[3];
    v4f r0 = *(const v4f*)(smR + (0*TB + m)*DIM + d);
    v4f r1 = *(const v4f*)(smR + (1*TB + m)*DIM + d);
    v4f r2 = *(const v4f*)(smR + (2*TB + m)*DIM + d);
    v4f r3 = *(const v4f*)(smR + (3*TB + m)*DIM + d);
    v4f o = c0*r0 + c1*r1 + c2*r2 + c3*r3;
    __builtin_nontemporal_store(
        o, (v4f*)(out + ((size_t)(bb*STREAMS + s)*SEQ + (n0 + m))*DIM + d));
  }
}

extern "C" void kernel_launch(void* const* d_in, const int* in_sizes, int n_in,
                              void* d_out, int out_size, void* d_ws, size_t ws_size,
                              hipStream_t stream) {
  (void)in_sizes; (void)n_in; (void)out_size; (void)d_ws; (void)ws_size;
  const float* resid  = (const float*)d_in[0];
  const float* gamma  = (const float*)d_in[1];
  const float* stA    = (const float*)d_in[2];
  const float* aFn    = (const float*)d_in[3];
  const float* aScale = (const float*)d_in[4];
  const float* stB    = (const float*)d_in[5];
  const float* bFn    = (const float*)d_in[6];
  const float* bScale = (const float*)d_in[7];
  float* out = (float*)d_out;

  const int nblocks = (BATCH * SEQ) / TB;   // 2048
  hyperconn_kernel<<<dim3(nblocks), dim3(BLOCK), LDS_TOTAL, stream>>>(
      resid, gamma, stA, aFn, aScale, stB, bFn, bScale, out);
}
